// graphCNN_5162550689990
// MI455X (gfx1250) — matrix operand order repaired and checked by reading, not compiled, not run on hardware
//
#include <hip/hip_runtime.h>

typedef __attribute__((ext_vector_type(16))) _Float16 v16h;
typedef __attribute__((ext_vector_type(8)))  _Float16 v8h;
typedef __attribute__((ext_vector_type(8)))  float    v8f;

__device__ __forceinline__ v16h joinHalves(v8h lo, v8h hi) {
  return __builtin_shufflevector(lo, hi, 0,1,2,3,4,5,6,7,8,9,10,11,12,13,14,15);
}

__device__ __forceinline__ void storeTile(float* __restrict__ C, int N, int m0, int n0,
                                          int lr, int hb, v8f acc) {
  union { v8f v; float f[8]; } u; u.v = acc;
  float* cp = C + (size_t)(m0 + (hb << 3)) * N + n0 + lr;
  #pragma unroll
  for (int r = 0; r < 8; ++r) cp[(size_t)r * N] = u.f[r];
}

// ---------------------------------------------------------------------------
// Register-blocked WMMA GEMM: C[M,N] = A[M,K] * W[N,K]^T ; A,W f16 row-major,
// C f32. Each wave computes a 32x32 C block (2x2 WMMA tiles), K in chunks of
// 32 via v_wmma_f32_16x16x32_f16: 8 b128 loads feed 4 WMMAs (2x fragment
// reuse vs the naive 1-tile version). M,N must be multiples of 32.
// ---------------------------------------------------------------------------
__global__ __launch_bounds__(256) void wmma_gemm_f16_2x2(
    const _Float16* __restrict__ A, const _Float16* __restrict__ W,
    float* __restrict__ C, int M, int N, int K)
{
  const int lane = threadIdx.x & 31;
  const int wave = threadIdx.x >> 5;
  const int nTiles = N >> 5;
  const int totalTiles = (M >> 5) * nTiles;
  const int tile = blockIdx.x * 8 + wave;
  if (tile >= totalTiles) return;
  const int mt = tile / nTiles, nt = tile - mt * nTiles;
  const int m0 = mt << 5, n0 = nt << 5;
  const int lr = lane & 15;   // A-row / B-col within 16x16 tile
  const int hb = lane >> 4;   // lane-half selector
  const _Float16* ap0 = A + (size_t)(m0 + lr) * K + hb * 8;
  const _Float16* ap1 = ap0 + (size_t)16 * K;
  const _Float16* bp0 = W + (size_t)(n0 + lr) * K + hb * 8;
  const _Float16* bp1 = bp0 + (size_t)16 * K;
  v8f acc00 = {}, acc01 = {}, acc10 = {}, acc11 = {};
  for (int k0 = 0; k0 < K; k0 += 32) {
    v16h a0 = joinHalves(*(const v8h*)(ap0 + k0), *(const v8h*)(ap0 + k0 + 16));
    v16h a1 = joinHalves(*(const v8h*)(ap1 + k0), *(const v8h*)(ap1 + k0 + 16));
    v16h b0 = joinHalves(*(const v8h*)(bp0 + k0), *(const v8h*)(bp0 + k0 + 16));
    v16h b1 = joinHalves(*(const v8h*)(bp1 + k0), *(const v8h*)(bp1 + k0 + 16));
    acc00 = __builtin_amdgcn_wmma_f32_16x16x32_f16(false, a0, false, b0, (short)0, acc00, false, false);
    acc01 = __builtin_amdgcn_wmma_f32_16x16x32_f16(false, a0, false, b1, (short)0, acc01, false, false);
    acc10 = __builtin_amdgcn_wmma_f32_16x16x32_f16(false, a1, false, b0, (short)0, acc10, false, false);
    acc11 = __builtin_amdgcn_wmma_f32_16x16x32_f16(false, a1, false, b1, (short)0, acc11, false, false);
  }
  storeTile(C, N, m0,      n0,      lr, hb, acc00);
  storeTile(C, N, m0,      n0 + 16, lr, hb, acc01);
  storeTile(C, N, m0 + 16, n0,      lr, hb, acc10);
  storeTile(C, N, m0 + 16, n0 + 16, lr, hb, acc11);
}

// ---------------------------------------------------------------------------
// Per-branch weight folding: conv1+BN1 -> affine; conv2 collapses to a
// single-input-channel 3x3 (we2); BN2/pool/bias fold into conv3 weights
// (W3f, f16) and bias (B3f).
// ---------------------------------------------------------------------------
struct BParams {
  const float *w1,*b1,*g1,*be1,*m1,*v1;
  const float *w2,*b2,*g2,*be2,*m2,*v2;
  const float *w3,*b3,*g3,*be3,*m3,*v3;
};

__global__ __launch_bounds__(256) void prep_branch(BParams P, float* __restrict__ we2,
                                                   _Float16* __restrict__ w3f,
                                                   float* __restrict__ b3f)
{
  __shared__ float Af[8], Cf[8], s2s[32], d2s[32];
  const int tid = threadIdx.x;
  if (tid < 8) {
    float s1 = P.g1[tid] * rsqrtf(P.v1[tid] + 1e-5f);
    Af[tid] = P.w1[tid] * s1;
    Cf[tid] = (P.b1[tid] - P.m1[tid]) * s1 + P.be1[tid];
  }
  __syncthreads();
  for (int i = tid; i < 288; i += 256) {          // we2[32][9]
    int o = i / 9, r = i - o * 9;
    float s = 0.f;
    #pragma unroll
    for (int c = 0; c < 8; ++c) s += P.w2[(o*8 + c)*9 + r] * Af[c];
    we2[i] = s;
  }
  if (tid < 32) {
    float bb = P.b2[tid];
    for (int c = 0; c < 8; ++c)
      for (int r = 0; r < 9; ++r) bb += P.w2[(tid*8 + c)*9 + r] * Cf[c];
    float s2 = P.g2[tid] * rsqrtf(P.v2[tid] + 1e-5f);
    s2s[tid] = s2;
    d2s[tid] = s2 * (bb - P.m2[tid]) + P.be2[tid];
  }
  __syncthreads();
  for (int i = tid; i < 64*288; i += 256) {       // W3f[64][288] f16
    int u = i / 288, k = i - u * 288, c = k / 9;
    float s3 = P.g3[u] * rsqrtf(P.v3[u] + 1e-5f);
    w3f[i] = (_Float16)(s3 * s2s[c] * P.w3[i]);
  }
  if (tid < 64) {
    float acc = P.b3[tid] - P.m3[tid];
    for (int k = 0; k < 288; ++k) acc += P.w3[tid*288 + k] * d2s[k / 9];
    float s3 = P.g3[tid] * rsqrtf(P.v3[tid] + 1e-5f);
    b3f[tid] = s3 * acc + P.be3[tid];
  }
}

// ---------------------------------------------------------------------------
// Fused branch pipeline: one block per (batch*month, branch) image.
// folded-conv2 + 3x3/3 maxpool (VALU) -> conv3 as implicit-im2col WMMA GEMM:
// each wave owns one M-tile (16 output pixels), gathers its A fragment ONCE
// per K-chunk and sweeps all 4 N-tiles (64 channels) with it (4x fewer LDS
// gathers than tile-per-wave) -> 2x2 maxpool + ReLU -> f16 feature.
// ---------------------------------------------------------------------------
__global__ __launch_bounds__(256) void branch_kernel(
    const float* __restrict__ sst, const float* __restrict__ t300,
    const float* __restrict__ ua,  const float* __restrict__ va,
    const float* __restrict__ we2_all, const _Float16* __restrict__ w3f_all,
    const float* __restrict__ b3f_all, _Float16* __restrict__ F)
{
  const int bt = blockIdx.x >> 2;
  const int br = blockIdx.x & 3;
  const float* xin = (br == 0) ? sst : (br == 1) ? t300 : (br == 2) ? ua : va;
  xin += (size_t)bt * (24 * 72);
  const float*    we2 = we2_all + br * 288;
  const _Float16* w3f = w3f_all + (size_t)br * (64 * 288);
  const float*    b3f = b3f_all + br * 64;

  __shared__ float    xs[24 * 72];      // input image
  __shared__ float    we2s[288];        // folded conv2 weights
  __shared__ _Float16 ps[32][7][23];    // pooled conv2 output (pre-BN2, folded)
  __shared__ float    outs[64][112];    // conv3 output [channel][pixel]

  const int tid = threadIdx.x;
  for (int i = tid; i < 24 * 72; i += 256) xs[i] = xin[i];
  for (int i = tid; i < 288; i += 256) we2s[i] = we2[i];
  __syncthreads();

  // folded conv2 (1ch 3x3) fused with 3x3 stride-3 maxpool
  for (int i = tid; i < 32 * 7 * 23; i += 256) {
    int c = i / 161, rem = i - c * 161, py = rem / 23, px = rem - py * 23;
    float w[9];
    #pragma unroll
    for (int r = 0; r < 9; ++r) w[r] = we2s[c * 9 + r];
    float mx = -3.4e38f;
    #pragma unroll
    for (int dy = 0; dy < 3; ++dy)
      #pragma unroll
      for (int dx = 0; dx < 3; ++dx) {
        int oy = 3 * py + dy, ox = 3 * px + dx;
        float s = 0.f;
        #pragma unroll
        for (int ky = 0; ky < 3; ++ky)
          #pragma unroll
          for (int kx = 0; kx < 3; ++kx)
            s += w[ky * 3 + kx] * xs[(oy + ky) * 72 + ox + kx];
        mx = fmaxf(mx, s);
      }
    ps[c][py][px] = (_Float16)mx;
  }
  __syncthreads();

  // conv3: D[105(pad112) x 64] = im2col(ps)[112 x 288] * W3f[64 x 288]^T
  const int lane = tid & 31, wave = tid >> 5;
  const int lr = lane & 15, hb = lane >> 4;
  if (wave < 7) {                       // wave w owns M-tile w (pixels 16w..16w+15)
    const int m0 = wave << 4;
    const int pix = m0 + lr;
    const bool valid = pix < 105;
    const int pixc = valid ? pix : 0;
    const int oy = pixc / 21, ox = pixc - oy * 21;
    v8f acc[4] = {{}, {}, {}, {}};
    #pragma unroll
    for (int kc = 0; kc < 288; kc += 32) {
      union { v16h v; _Float16 e[16]; } af;
      #pragma unroll
      for (int h = 0; h < 16; ++h) {
        int k = kc + hb * 16 + h;   // contiguous A-fragment K map (matches B)
        int c = k / 9, r = k - c * 9, ky = r / 3, kx = r - ky * 3;
        _Float16 val = ps[c][oy + ky][ox + kx];
        af.e[h] = valid ? val : (_Float16)0.f;
      }
      #pragma unroll
      for (int nt = 0; nt < 4; ++nt) {  // reuse A fragment over all 4 N-tiles
        const _Float16* bp = w3f + (size_t)((nt << 4) + lr) * 288 + hb * 16 + kc;
        v16h bv = joinHalves(*(const v8h*)bp, *(const v8h*)(bp + 8));
        acc[nt] = __builtin_amdgcn_wmma_f32_16x16x32_f16(false, af.v, false, bv,
                                                         (short)0, acc[nt], false, false);
      }
    }
    #pragma unroll
    for (int nt = 0; nt < 4; ++nt) {
      union { v8f v; float f[8]; } u; u.v = acc[nt];
      const int ch = (nt << 4) + lr;
      const float bias = b3f[ch];
      #pragma unroll
      for (int r = 0; r < 8; ++r)
        outs[ch][m0 + (hb << 3) + r] = u.f[r] + bias;
    }
  }
  __syncthreads();

  // 2x2 maxpool + ReLU, write feature [64*2*10] f16
  _Float16* fout = F + ((size_t)bt * 4 + br) * 1280;
  for (int i = tid; i < 1280; i += 256) {
    int u2 = i / 20, rem = i - u2 * 20, yy = rem / 10, xx = rem - yy * 10;
    int p0 = (2 * yy) * 21 + 2 * xx;
    float a = fmaxf(fmaxf(outs[u2][p0],      outs[u2][p0 + 1]),
                    fmaxf(outs[u2][p0 + 21], outs[u2][p0 + 22]));
    fout[i] = (_Float16)fmaxf(a, 0.f);
  }
}

// f32 -> f16 elementwise
__global__ void f32_to_f16(const float* __restrict__ src, _Float16* __restrict__ dst, int n) {
  int i = blockIdx.x * blockDim.x + threadIdx.x;
  if (i < n) dst[i] = (_Float16)src[i];
}

// X[bt,n,j] = f16( bias[j] + sum_m adj[n,m] * G[bt*4+m, j] )
__global__ void gcn_mix(const float* __restrict__ G, const float* __restrict__ adj,
                        const float* __restrict__ bias, _Float16* __restrict__ X,
                        int Fdim, int total)
{
  int i = blockIdx.x * blockDim.x + threadIdx.x;
  if (i >= total) return;
  int j = i % Fdim;
  int n = (i / Fdim) & 3;
  int bt = i / (Fdim * 4);
  float s = bias[j];
  #pragma unroll
  for (int m = 0; m < 4; ++m) s += adj[n * 4 + m] * G[((size_t)bt * 4 + m) * Fdim + j];
  X[i] = (_Float16)s;
}

// ---------------------------------------------------------------------------
// LSTM recurrence (reference scans axis 0: T=128 steps, batch N=12, H=64).
// One block per direction; whh row held in VGPRs, h/c/gates in LDS.
// U = x @ Wih^T precomputed via WMMA GEMM; biases added here.
// ---------------------------------------------------------------------------
__global__ __launch_bounds__(256) void lstm_recur(
    const float* __restrict__ U0, const float* __restrict__ U1,
    const float* __restrict__ whh0, const float* __restrict__ whh1,
    const float* __restrict__ bih0, const float* __restrict__ bhh0,
    const float* __restrict__ bih1, const float* __restrict__ bhh1,
    _Float16* __restrict__ out)
{
  const int dir = blockIdx.x;
  const float* U   = dir ? U1   : U0;
  const float* whh = dir ? whh1 : whh0;
  const float* bih = dir ? bih1 : bih0;
  const float* bhh = dir ? bhh1 : bhh0;
  __shared__ float hs[12][64], cs[12][64], gs[12][256];
  const int tid = threadIdx.x;
  for (int i = tid; i < 12 * 64; i += 256) { hs[i >> 6][i & 63] = 0.f; cs[i >> 6][i & 63] = 0.f; }
  float wrow[64];
  #pragma unroll
  for (int k = 0; k < 64; ++k) wrow[k] = whh[tid * 64 + k];
  const float bsum = bih[tid] + bhh[tid];
  for (int t = 0; t < 128; ++t) {
    const int step = dir ? (127 - t) : t;
    __syncthreads();
    for (int n = 0; n < 12; ++n) {
      float g = U[((size_t)step * 12 + n) * 256 + tid] + bsum;
      #pragma unroll
      for (int k = 0; k < 64; ++k) g += wrow[k] * hs[n][k];
      gs[n][tid] = g;
    }
    __syncthreads();
    for (int cell = tid; cell < 768; cell += 256) {
      int n = cell >> 6, k = cell & 63;
      float ig = 1.f / (1.f + __expf(-gs[n][k]));
      float fg = 1.f / (1.f + __expf(-gs[n][64 + k]));
      float gg = tanhf(gs[n][128 + k]);
      float og = 1.f / (1.f + __expf(-gs[n][192 + k]));
      float cn = fg * cs[n][k] + ig * gg;
      cs[n][k] = cn;
      float hn = og * tanhf(cn);
      hs[n][k] = hn;
      out[((size_t)step * 12 + n) * 128 + dir * 64 + k] = (_Float16)hn;
    }
  }
}

// mean over months + final 128->24 linear
__global__ __launch_bounds__(128) void head_kernel(
    const _Float16* __restrict__ hc, const float* __restrict__ lw,
    const float* __restrict__ lb, float* __restrict__ out)
{
  __shared__ float m[128];
  const int b = blockIdx.x, k = threadIdx.x;
  float s = 0.f;
  for (int t = 0; t < 12; ++t) s += (float)hc[((size_t)b * 12 + t) * 128 + k];
  m[k] = s * (1.f / 12.f);
  __syncthreads();
  if (k < 24) {
    float a = lb[k];
    #pragma unroll
    for (int kk = 0; kk < 128; ++kk) a += lw[k * 128 + kk] * m[kk];
    out[b * 24 + k] = a;
  }
}

// ---------------------------------------------------------------------------
extern "C" void kernel_launch(void* const* d_in, const int* in_sizes, int n_in,
                              void* d_out, int out_size, void* d_ws, size_t ws_size,
                              hipStream_t stream)
{
  (void)in_sizes; (void)n_in; (void)out_size; (void)ws_size;
  const float* sst    = (const float*)d_in[0];
  const float* t300   = (const float*)d_in[1];
  const float* ua     = (const float*)d_in[2];
  const float* va     = (const float*)d_in[3];
  const float* adj    = (const float*)d_in[4];
  const float* gcn1_w = (const float*)d_in[77];
  const float* gcn1_b = (const float*)d_in[78];
  const float* gcn2_w = (const float*)d_in[79];
  const float* gcn2_b = (const float*)d_in[80];
  const float* wih_l0  = (const float*)d_in[81];
  const float* whh_l0  = (const float*)d_in[82];
  const float* bih_l0  = (const float*)d_in[83];
  const float* bhh_l0  = (const float*)d_in[84];
  const float* wih_l0r = (const float*)d_in[85];
  const float* whh_l0r = (const float*)d_in[86];
  const float* bih_l0r = (const float*)d_in[87];
  const float* bhh_l0r = (const float*)d_in[88];
  const float* wih_l1  = (const float*)d_in[89];
  const float* whh_l1  = (const float*)d_in[90];
  const float* bih_l1  = (const float*)d_in[91];
  const float* bhh_l1  = (const float*)d_in[92];
  const float* wih_l1r = (const float*)d_in[93];
  const float* whh_l1r = (const float*)d_in[94];
  const float* bih_l1r = (const float*)d_in[95];
  const float* bhh_l1r = (const float*)d_in[96];
  const float* lin_w  = (const float*)d_in[97];
  const float* lin_b  = (const float*)d_in[98];

  char* ws = (char*)d_ws;
  size_t off = 0;
  auto alloc = [&](size_t bytes) -> void* {
    void* p = (void*)(ws + off);
    off = (off + bytes + 255) & ~(size_t)255;
    return p;
  };
  float*     we2_ws = (float*)    alloc(4 * 288 * sizeof(float));
  _Float16*  w3f_ws = (_Float16*) alloc((size_t)4 * 64 * 288 * sizeof(_Float16));
  float*     b3f_ws = (float*)    alloc(4 * 64 * sizeof(float));
  _Float16*  g1w16  = (_Float16*) alloc((size_t)256 * 1280 * sizeof(_Float16));
  _Float16*  g2w16  = (_Float16*) alloc((size_t)64 * 256 * sizeof(_Float16));
  _Float16*  wih0f  = (_Float16*) alloc((size_t)256 * 256 * sizeof(_Float16));
  _Float16*  wih0r  = (_Float16*) alloc((size_t)256 * 256 * sizeof(_Float16));
  _Float16*  wih1f  = (_Float16*) alloc((size_t)256 * 128 * sizeof(_Float16));
  _Float16*  wih1r  = (_Float16*) alloc((size_t)256 * 128 * sizeof(_Float16));
  _Float16*  F16    = (_Float16*) alloc((size_t)6144 * 1280 * sizeof(_Float16));
  float*     G      = (float*)    alloc((size_t)6144 * 256 * sizeof(float));
  _Float16*  X1     = (_Float16*) alloc((size_t)6144 * 256 * sizeof(_Float16));
  float*     Hbuf   = (float*)    alloc((size_t)6144 * 64 * sizeof(float));
  _Float16*  XIN    = (_Float16*) alloc((size_t)6144 * 64 * sizeof(_Float16)); // = [1536,256]
  float*     U0     = (float*)    alloc((size_t)1536 * 256 * sizeof(float));
  float*     U1     = (float*)    alloc((size_t)1536 * 256 * sizeof(float));
  _Float16*  HC0    = (_Float16*) alloc((size_t)1536 * 128 * sizeof(_Float16));
  _Float16*  HC1    = (_Float16*) alloc((size_t)1536 * 128 * sizeof(_Float16));

  // 1) fold BN/conv weights per branch
  for (int br = 0; br < 4; ++br) {
    const int base = 5 + 18 * br;
    BParams P;
    P.w1  = (const float*)d_in[base + 0];  P.b1  = (const float*)d_in[base + 1];
    P.g1  = (const float*)d_in[base + 2];  P.be1 = (const float*)d_in[base + 3];
    P.m1  = (const float*)d_in[base + 4];  P.v1  = (const float*)d_in[base + 5];
    P.w2  = (const float*)d_in[base + 6];  P.b2  = (const float*)d_in[base + 7];
    P.g2  = (const float*)d_in[base + 8];  P.be2 = (const float*)d_in[base + 9];
    P.m2  = (const float*)d_in[base + 10]; P.v2  = (const float*)d_in[base + 11];
    P.w3  = (const float*)d_in[base + 12]; P.b3  = (const float*)d_in[base + 13];
    P.g3  = (const float*)d_in[base + 14]; P.be3 = (const float*)d_in[base + 15];
    P.m3  = (const float*)d_in[base + 16]; P.v3  = (const float*)d_in[base + 17];
    prep_branch<<<1, 256, 0, stream>>>(P, we2_ws + br * 288,
                                       w3f_ws + (size_t)br * 64 * 288,
                                       b3f_ws + br * 64);
  }

  // 2) f16 weight copies for WMMA GEMMs
  auto cvt = [&](const float* s, _Float16* d, int n) {
    f32_to_f16<<<(n + 255) / 256, 256, 0, stream>>>(s, d, n);
  };
  cvt(gcn1_w, g1w16, 256 * 1280);
  cvt(gcn2_w, g2w16, 64 * 256);
  cvt(wih_l0,  wih0f, 256 * 256);
  cvt(wih_l0r, wih0r, 256 * 256);
  cvt(wih_l1,  wih1f, 256 * 128);
  cvt(wih_l1r, wih1r, 256 * 128);

  // 3) fused branch CNN (4 branches x 1536 images)
  branch_kernel<<<6144, 256, 0, stream>>>(sst, t300, ua, va,
                                          we2_ws, w3f_ws, b3f_ws, F16);

  auto gemm = [&](const _Float16* A, const _Float16* W, float* C, int M, int N, int K) {
    int tiles = (M / 32) * (N / 32);
    wmma_gemm_f16_2x2<<<(tiles + 7) / 8, 256, 0, stream>>>(A, W, C, M, N, K);
  };

  // 4) GCN1: G = F @ W1^T ; X1 = adj-mix(G) + b1
  gemm(F16, g1w16, G, 6144, 256, 1280);
  gcn_mix<<<(6144 * 256 + 255) / 256, 256, 0, stream>>>(G, adj, gcn1_b, X1, 256, 6144 * 256);

  // 5) GCN2: H = X1 @ W2^T ; XIN = adj-mix(H) + b2  (== LSTM input [1536,256])
  gemm(X1, g2w16, Hbuf, 6144, 64, 256);
  gcn_mix<<<(6144 * 64 + 255) / 256, 256, 0, stream>>>(Hbuf, adj, gcn2_b, XIN, 64, 6144 * 64);

  // 6) BiLSTM layer 0 (T=128, N=12 per reference's axis-0 scan)
  gemm(XIN, wih0f, U0, 1536, 256, 256);
  gemm(XIN, wih0r, U1, 1536, 256, 256);
  lstm_recur<<<2, 256, 0, stream>>>(U0, U1, whh_l0, whh_l0r,
                                    bih_l0, bhh_l0, bih_l0r, bhh_l0r, HC0);

  // 7) BiLSTM layer 1
  gemm(HC0, wih1f, U0, 1536, 256, 128);
  gemm(HC0, wih1r, U1, 1536, 256, 128);
  lstm_recur<<<2, 256, 0, stream>>>(U0, U1, whh_l1, whh_l1r,
                                    bih_l1, bhh_l1, bih_l1r, bhh_l1r, HC1);

  // 8) month-mean + final linear
  head_kernel<<<128, 128, 0, stream>>>(HC1, lin_w, lin_b, (float*)d_out);
}
